// LinearAttention_46952582480083
// MI455X (gfx1250) — matrix operand order repaired and checked by reading, not compiled, not run on hardware
//
#include <hip/hip_runtime.h>

// ---------------------------------------------------------------------------
// LinearAttention forward for MI455X (gfx1250, wave32, WMMA bf16).
// Pipeline: wconv -> layernorm -> GEMM(qkv) -> softmax(k) -> GEMM(ctx)
//           -> GEMM(out_heads) -> GEMM(final + bias + residual)
// Matmuls use v_wmma_f32_16x16x32_bf16 (f32 accumulate).
// CDNA5 data movers: tensor_load_to_lds (TDM, TENSORcnt) stages the ctx-GEMM
// tiles; global_load_async_to_lds_b128 (ASYNCcnt) stages the A tiles of the
// big GEMMs.
// ---------------------------------------------------------------------------

typedef __attribute__((ext_vector_type(16))) __bf16 v16bf;
typedef __attribute__((ext_vector_type(8)))  float  v8f;
typedef __attribute__((ext_vector_type(4)))  unsigned int v4u;
typedef __attribute__((ext_vector_type(8)))  unsigned int v8u;

#define BATCH 16
#define DCH   384     // channel dim D
#define SEQ   8192    // sequence length N
#define NHEAD 8
#define DHEAD 64
#define HTOT  512     // NHEAD*DHEAD
#define H3    1536    // 3*HTOT
#define QSCALE 0.125f // DHEAD^-0.5
#define LN_EPS 1e-5f

static __device__ __forceinline__ v8f wmma_bf16(v16bf a, v16bf b, v8f c) {
  return __builtin_amdgcn_wmma_f32_16x16x32_bf16(
      /*neg_a=*/false, a, /*neg_b=*/false, b,
      /*c_mod=*/(short)0, c, /*reuse_a=*/false, /*reuse_b=*/false);
}

// A-matrix fragment (16x32, K contiguous in LDS row):
// lane<16: row=lane, K {0..7, 16..23}; lane>=16: row=lane-16, K {8..15, 24..31}
static __device__ __forceinline__ v16bf frag_a(const __bf16* rowp, int lane) {
  union { v16bf v; uint4 q[2]; } u;
  const int kb = (lane >> 4) * 8;
  u.q[0] = *(const uint4*)(rowp + kb);
  u.q[1] = *(const uint4*)(rowp + kb + 16);
  return u.v;
}
// B-matrix fragment (32x16) from transposed LDS tile Bs[col][k]:
// lane<16: col=lane, K {0..7, 16..23}; lane>=16: col=lane-16, K {8..15, 24..31}
static __device__ __forceinline__ v16bf frag_b(const __bf16* colp, int lane) {
  union { v16bf v; uint4 q[2]; } u;
  const int kb = (lane >> 4) * 8;
  u.q[0] = *(const uint4*)(colp + kb);
  u.q[1] = *(const uint4*)(colp + kb + 16);
  return u.v;
}

// ---------------------------------------------------------------------------
// Weight conversion fp32 -> bf16 (weights reused SEQ times; convert once)
// ---------------------------------------------------------------------------
__global__ __launch_bounds__(256)
void la_wconv(const float* __restrict__ wq, const float* __restrict__ wo,
              __bf16* __restrict__ wqb, __bf16* __restrict__ wob) {
  const int i = blockIdx.x * 256 + threadIdx.x;
  const int n1 = H3 * DCH;
  if (i < n1) wqb[i] = (__bf16)wq[i];
  const int j = i - n1;
  if (j >= 0 && j < DCH * HTOT) wob[j] = (__bf16)wo[j];
}

// ---------------------------------------------------------------------------
// LayerNorm over D per (b,n); writes bf16 x_norm. Coalesced along n.
// ---------------------------------------------------------------------------
__global__ __launch_bounds__(256)
void la_layernorm(const float* __restrict__ x, const float* __restrict__ g,
                  const float* __restrict__ be, __bf16* __restrict__ xn) {
  const int n = blockIdx.x * 256 + threadIdx.x;
  const int b = blockIdx.y;
  const float* xp = x + (size_t)b * DCH * SEQ + n;
  float s = 0.f, sq = 0.f;
  for (int d = 0; d < DCH; ++d) {
    float v = xp[(size_t)d * SEQ];
    s += v; sq += v * v;
  }
  const float mean = s * (1.f / DCH);
  const float var  = sq * (1.f / DCH) - mean * mean;
  const float rstd = rsqrtf(var + LN_EPS);
  __bf16* op = xn + (size_t)b * DCH * SEQ + n;
  for (int d = 0; d < DCH; ++d) {
    float v = (xp[(size_t)d * SEQ] - mean) * rstd * g[d] + be[d];
    op[(size_t)d * SEQ] = (__bf16)v;
  }
}

// ---------------------------------------------------------------------------
// Tiled bf16 GEMM: C(MxN) = A(MxK) * B(KxN), batched over blockIdx.z on B/C.
// Block tile 128x128, K-step 32, 8 waves arranged 4(M) x 2(N);
// each wave owns 32x64 = 2x4 WMMA tiles.
// A tile is staged with CDNA5 async global->LDS copies (ASYNCcnt).
// MODE 0: store bf16 (qkv).  MODE 1: + bias[row] + residual, store fp32.
// ---------------------------------------------------------------------------
template <int MODE>
__global__ __launch_bounds__(256)
void la_gemm128(const __bf16* __restrict__ A, const __bf16* __restrict__ Bm,
                void* __restrict__ Cout, const float* __restrict__ bias,
                const float* __restrict__ resid, int M, int K, int N) {
  __shared__ alignas(16) __bf16 As[128][40];   // [m][k], padded rows
  __shared__ alignas(16) __bf16 Bs[128][40];   // [n][k] (transposed on store)
  const int tid = threadIdx.x, lane = tid & 31, wid = tid >> 5;
  const int wr = wid & 3, wc = wid >> 2;
  const int mB = blockIdx.y * 128, nB = blockIdx.x * 128;
  const int bz = blockIdx.z;
  Bm += (size_t)bz * K * N;
  const int arow = tid >> 1, akc = (tid & 1) * 16;
  const unsigned alds = (unsigned)(size_t)&As[arow][akc];
  v8f acc[2][4] = {};
  for (int k0 = 0; k0 < K; k0 += 32) {
    // stage A tile 128x32 via async global->LDS (two B128 per thread)
    {
      const __bf16* gp = A + (size_t)(mB + arow) * K + k0 + akc;
      asm volatile(
          "global_load_async_to_lds_b128 %0, %1, off\n\t"
          "global_load_async_to_lds_b128 %2, %3, off"
          :: "v"(alds), "v"(gp), "v"(alds + 16u), "v"(gp + 8)
          : "memory");
    }
    // stage B tile 32x128 transposed into Bs[n][k]
#pragma unroll
    for (int it = 0; it < 2; ++it) {
      const int ch = tid + it * 256;        // 512 chunks of 8 elems
      const int kk = ch >> 4, n0 = (ch & 15) * 8;
      uint4 d = *(const uint4*)(Bm + (size_t)(k0 + kk) * N + nB + n0);
      const __bf16* e = (const __bf16*)&d;
#pragma unroll
      for (int j = 0; j < 8; ++j) Bs[n0 + j][kk] = e[j];
    }
    if (k0 + 32 < K)  // prefetch next B K-tile (global_prefetch_b8)
      __builtin_prefetch(Bm + (size_t)(k0 + 32 + (tid >> 4)) * N + nB + (tid & 15) * 8, 0, 0);
    asm volatile("s_wait_asynccnt 0x0" ::: "memory");
    __syncthreads();
    v16bf af[2], bfr[4];
#pragma unroll
    for (int mt = 0; mt < 2; ++mt)
      af[mt] = frag_a(&As[wr * 32 + mt * 16 + (lane & 15)][0], lane);
#pragma unroll
    for (int nt = 0; nt < 4; ++nt)
      bfr[nt] = frag_b(&Bs[wc * 64 + nt * 16 + (lane & 15)][0], lane);
#pragma unroll
    for (int mt = 0; mt < 2; ++mt)
#pragma unroll
      for (int nt = 0; nt < 4; ++nt)
        acc[mt][nt] = wmma_bf16(af[mt], bfr[nt], acc[mt][nt]);
    __syncthreads();
  }
  const int hi = (lane >> 4) * 8, ln = lane & 15;
#pragma unroll
  for (int mt = 0; mt < 2; ++mt) {
#pragma unroll
    for (int nt = 0; nt < 4; ++nt) {
      const int rbase = mB + wr * 32 + mt * 16 + hi;
      const int col   = nB + wc * 64 + nt * 16 + ln;
#pragma unroll
      for (int r = 0; r < 8; ++r) {
        const size_t idx = (size_t)(rbase + r) * N + col;
        if constexpr (MODE == 0) {
          ((__bf16*)Cout + (size_t)bz * M * N)[idx] = (__bf16)acc[mt][nt][r];
        } else {
          const float* rz = resid + (size_t)bz * M * N;
          ((float*)Cout + (size_t)bz * M * N)[idx] =
              acc[mt][nt][r] + bias[rbase + r] + rz[idx];
        }
      }
    }
  }
}

// ---------------------------------------------------------------------------
// Row softmax over SEQ for the k third of qkv, in place. One block per row.
// ---------------------------------------------------------------------------
__global__ __launch_bounds__(256)
void la_softmax_k(__bf16* __restrict__ qkv) {
  __shared__ float red[256];
  const int tid = threadIdx.x;
  const int r = blockIdx.x;            // 0 .. BATCH*HTOT-1
  const int b = r >> 9, rr = r & 511;
  __bf16* kp = qkv + ((size_t)b * H3 + HTOT + rr) * SEQ;
  float f[32];
  float m = -3.4e38f;
#pragma unroll
  for (int c = 0; c < 4; ++c) {
    union { uint4 q; __bf16 e[8]; } u;
    u.q = ((const uint4*)kp)[tid + c * 256];
#pragma unroll
    for (int j = 0; j < 8; ++j) {
      f[c * 8 + j] = (float)u.e[j];
      m = fmaxf(m, f[c * 8 + j]);
    }
  }
  red[tid] = m; __syncthreads();
  for (int s = 128; s > 0; s >>= 1) {
    if (tid < s) red[tid] = fmaxf(red[tid], red[tid + s]);
    __syncthreads();
  }
  m = red[0]; __syncthreads();
  float sum = 0.f;
#pragma unroll
  for (int i = 0; i < 32; ++i) { f[i] = __expf(f[i] - m); sum += f[i]; }
  red[tid] = sum; __syncthreads();
  for (int s = 128; s > 0; s >>= 1) {
    if (tid < s) red[tid] += red[tid + s];
    __syncthreads();
  }
  const float inv = 1.0f / red[0];
#pragma unroll
  for (int c = 0; c < 4; ++c) {
    union { uint4 q; __bf16 e[8]; } u;
#pragma unroll
    for (int j = 0; j < 8; ++j) u.e[j] = (__bf16)(f[c * 8 + j] * inv);
    ((uint4*)kp)[tid + c * 256] = u.q;
  }
}

// ---------------------------------------------------------------------------
// ctx[d,e] = sum_n k[d,n] * v[e,n] per (b,h). K-dim = SEQ = 8192.
// Tiles (64 rows x 32 bf16, row stride SEQ) are pulled into LDS by the
// Tensor Data Mover: one 2-D D# per tile, LDS row padding (64B data + 16B
// pad) done by TDM pad_enable so the tile lands directly in the padded
// [64][40] layout. Wave 0 issues the DMA; s_wait_tensorcnt + barrier sync.
// Output stored TRANSPOSED and pre-scaled: ctxT[e][d] = ctx[d][e] * QSCALE.
// ---------------------------------------------------------------------------
__global__ __launch_bounds__(256)
void la_gemm_ctx(const __bf16* __restrict__ qkv, __bf16* __restrict__ ctxT) {
  __shared__ alignas(16) __bf16 Ks[64][40];
  __shared__ alignas(16) __bf16 Vs[64][40];
  const int tid = threadIdx.x, lane = tid & 31, wid = tid >> 5;
  const int mt = wid & 3, ntp = wid >> 2;     // 4 d-tiles x 2 e-tile pairs
  const int bh = blockIdx.x, b = bh >> 3, h = bh & 7;
  const __bf16* kp = qkv + ((size_t)b * H3 + HTOT + h * DHEAD) * SEQ;
  const __bf16* vp = qkv + ((size_t)b * H3 + 2 * HTOT + h * DHEAD) * SEQ;

  // D# group1 (invariant): data_size=2B, pad_enable, pad_interval=16 dwords
  // (code 3), pad_amount=4 dwords (code 3); tensor_dim0=SEQ, tensor_dim1=64,
  // tile_dim0=32, tile_dim1=64, tensor_dim0_stride=SEQ.
  v8u g1;
  g1[0] = (1u << 16) | (1u << 20) | (3u << 22) | (3u << 25);
  g1[1] = ((unsigned)SEQ & 0xFFFFu) << 16;                    // dim0 lo
  g1[2] = (((unsigned)SEQ >> 16) & 0xFFFFu) | (64u << 16);    // dim0 hi|dim1 lo
  g1[3] = (32u << 16);                                        // tile_dim0
  g1[4] = 64u;                                                // tile_dim1
  g1[5] = (unsigned)SEQ;                                      // dim0_stride lo
  g1[6] = 0u;
  g1[7] = 0u;
  const unsigned ks_lds = (unsigned)(size_t)&Ks[0][0];
  const unsigned vs_lds = (unsigned)(size_t)&Vs[0][0];

  v8f acc[2] = {};
  for (int n0 = 0; n0 < SEQ; n0 += 32) {
    __syncthreads();                       // LDS free before DMA overwrites
    if (wid == 0) {
      const unsigned long long gk = (unsigned long long)(size_t)(kp + n0);
      const unsigned long long gv = (unsigned long long)(size_t)(vp + n0);
      v4u kg0, vg0;
      kg0[0] = 1u; kg0[1] = ks_lds;        // count=1, lds_addr
      kg0[2] = (unsigned)gk;
      kg0[3] = ((unsigned)(gk >> 32) & 0x01FFFFFFu) | 0x80000000u;  // type=2
      vg0[0] = 1u; vg0[1] = vs_lds;
      vg0[2] = (unsigned)gv;
      vg0[3] = ((unsigned)(gv >> 32) & 0x01FFFFFFu) | 0x80000000u;
      asm volatile("tensor_load_to_lds %0, %1" :: "s"(kg0), "s"(g1) : "memory");
      asm volatile("tensor_load_to_lds %0, %1" :: "s"(vg0), "s"(g1) : "memory");
      __builtin_amdgcn_s_wait_tensorcnt(0);
    }
    __syncthreads();                       // tiles visible to all waves
    v16bf af = frag_a(&Ks[mt * 16 + (lane & 15)][0], lane);
#pragma unroll
    for (int j = 0; j < 2; ++j) {
      v16bf bfj = frag_b(&Vs[(ntp * 2 + j) * 16 + (lane & 15)][0], lane);
      acc[j] = wmma_bf16(af, bfj, acc[j]);
    }
  }
  const int hi = (lane >> 4) * 8, ln = lane & 15;
  __bf16* cp = ctxT + (size_t)bh * DHEAD * DHEAD;
#pragma unroll
  for (int j = 0; j < 2; ++j) {
    const int e = (ntp * 2 + j) * 16 + ln;
#pragma unroll
    for (int r = 0; r < 8; ++r) {
      const int d = mt * 16 + r + hi;
      cp[e * DHEAD + d] = (__bf16)(acc[j][r] * QSCALE);  // q-scale folded here
    }
  }
}

// ---------------------------------------------------------------------------
// out_h[e,n] = sum_d ctxT[e,d] * q[d,n] per (b,h). M=64, K=64, N=SEQ.
// Block tile 64x128; 8 waves = 4(M) x 2(N-groups of 4 tiles).
// ---------------------------------------------------------------------------
__global__ __launch_bounds__(256)
void la_gemm_oh(const __bf16* __restrict__ qkv, const __bf16* __restrict__ ctxT,
                __bf16* __restrict__ outh) {
  __shared__ alignas(16) __bf16 Cs[64][72];   // ctxT tile (whole A)
  __shared__ alignas(16) __bf16 Bs[128][40];  // q tile transposed [n][k]
  const int tid = threadIdx.x, lane = tid & 31, wid = tid >> 5;
  const int mt = wid & 3, ng = wid >> 2;
  const int bh = blockIdx.y, b = bh >> 3, h = bh & 7;
  const int nB = blockIdx.x * 128;
  const __bf16* qp = qkv + ((size_t)b * H3 + h * DHEAD) * SEQ;
  const __bf16* cp = ctxT + (size_t)bh * DHEAD * DHEAD;
  {
    const int row = tid >> 2, c0 = (tid & 3) * 16;
    const uint4* s = (const uint4*)(cp + row * DHEAD + c0);
    *(uint4*)&Cs[row][c0]     = s[0];
    *(uint4*)&Cs[row][c0 + 8] = s[1];
  }
  v8f acc[4] = {};
  for (int k0 = 0; k0 < DHEAD; k0 += 32) {
#pragma unroll
    for (int it = 0; it < 2; ++it) {
      const int ch = tid + it * 256;
      const int kk = ch >> 4, n0 = (ch & 15) * 8;
      uint4 d = *(const uint4*)(qp + (size_t)(k0 + kk) * SEQ + nB + n0);
      const __bf16* e = (const __bf16*)&d;
#pragma unroll
      for (int j = 0; j < 8; ++j) Bs[n0 + j][kk] = e[j];
    }
    __syncthreads();
    v16bf af = frag_a(&Cs[mt * 16 + (lane & 15)][k0], lane);
#pragma unroll
    for (int j = 0; j < 4; ++j) {
      v16bf bfj = frag_b(&Bs[(ng * 4 + j) * 16 + (lane & 15)][0], lane);
      acc[j] = wmma_bf16(af, bfj, acc[j]);
    }
    __syncthreads();
  }
  const int hi = (lane >> 4) * 8, ln = lane & 15;
#pragma unroll
  for (int j = 0; j < 4; ++j) {
    const int col = nB + (ng * 4 + j) * 16 + ln;
#pragma unroll
    for (int r = 0; r < 8; ++r) {
      const int e = mt * 16 + r + hi;
      outh[((size_t)b * HTOT + h * DHEAD + e) * SEQ + col] = (__bf16)acc[j][r];
    }
  }
}

// ---------------------------------------------------------------------------
extern "C" void kernel_launch(void* const* d_in, const int* in_sizes, int n_in,
                              void* d_out, int out_size, void* d_ws, size_t ws_size,
                              hipStream_t stream) {
  (void)in_sizes; (void)n_in; (void)out_size; (void)ws_size;
  const float* x  = (const float*)d_in[0];
  const float* g  = (const float*)d_in[1];
  const float* be = (const float*)d_in[2];
  const float* wq = (const float*)d_in[3];
  const float* wo = (const float*)d_in[4];
  const float* bo = (const float*)d_in[5];
  float* out = (float*)d_out;

  char* ws = (char*)d_ws;
  size_t off = 0;
  auto take = [&](size_t bytes) -> char* {
    char* p = ws + off;
    off += (bytes + 255) & ~(size_t)255;
    return p;
  };
  __bf16* xnorm = (__bf16*)take((size_t)BATCH * DCH * SEQ * 2);              // ~100 MB
  __bf16* qkvb  = (__bf16*)take((size_t)BATCH * H3 * SEQ * 2);               // ~403 MB
  __bf16* ctxT  = (__bf16*)take((size_t)BATCH * NHEAD * DHEAD * DHEAD * 2);  // 1 MB
  __bf16* outh  = (__bf16*)take((size_t)BATCH * HTOT * SEQ * 2);             // ~134 MB
  __bf16* wqb   = (__bf16*)take((size_t)H3 * DCH * 2);
  __bf16* wob   = (__bf16*)take((size_t)DCH * HTOT * 2);

  la_wconv<<<dim3((H3 * DCH + DCH * HTOT + 255) / 256), 256, 0, stream>>>(wq, wo, wqb, wob);
  la_layernorm<<<dim3(SEQ / 256, BATCH), 256, 0, stream>>>(x, g, be, xnorm);
  la_gemm128<0><<<dim3(SEQ / 128, H3 / 128, BATCH), 256, 0, stream>>>(
      wqb, xnorm, qkvb, nullptr, nullptr, H3, DCH, SEQ);
  la_softmax_k<<<dim3(BATCH * HTOT), 256, 0, stream>>>(qkvb);
  la_gemm_ctx<<<dim3(BATCH * NHEAD), 256, 0, stream>>>(qkvb, ctxT);
  la_gemm_oh<<<dim3(SEQ / 128, BATCH * NHEAD), 256, 0, stream>>>(qkvb, ctxT, outh);
  la_gemm128<1><<<dim3(SEQ / 128, DCH / 128, BATCH), 256, 0, stream>>>(
      wob, outh, out, bo, x, DCH, HTOT, SEQ);
}
